// MLPResNetBlockPro_44066364457545
// MI455X (gfx1250) — matrix operand order repaired and checked by reading, not compiled, not run on hardware
//
#include <hip/hip_runtime.h>
#include <hip/hip_bf16.h>
#include <math.h>

#define B_    4
#define T_    1024
#define DIM_  1024
#define NH_   16
#define HD_   64
#define KA_   320
#define KT_   512
#define KTOT_ 1856          // 1024 tok + 320 ada + 512 tsk
#define SEG_ADA 1024
#define SEG_TSK 1344

typedef __bf16 bf16;
typedef __attribute__((ext_vector_type(16))) __bf16 bf16x16;
typedef __attribute__((ext_vector_type(8)))  __bf16 bf16x8;
typedef __attribute__((ext_vector_type(8)))  float  f32x8;
typedef __attribute__((ext_vector_type(4))) unsigned int u32x4;
typedef __attribute__((ext_vector_type(8))) int          i32x8;
typedef __attribute__((ext_vector_type(4))) int          i32x4;

static __device__ __forceinline__ bf16x8 load8(const bf16* p) {
  return *reinterpret_cast<const bf16x8*>(p);
}

// A-matrix (16x32 bf16) fragment: lane row = lane%16.
// element e -> K = kbase + half*16 + e
static __device__ __forceinline__ bf16x16 frag_A(const bf16* rowp, int kbase, int half) {
  bf16x8 lo = load8(rowp + kbase + half * 16);
  bf16x8 hi = load8(rowp + kbase + half * 16 + 8);
  bf16x16 r;
#pragma unroll
  for (int i = 0; i < 8; ++i) { r[i] = lo[i]; r[8 + i] = hi[i]; }
  return r;
}

// B-matrix (32x16 bf16) fragment: lane col = lane%16. element e -> K = kbase + half*16 + e
static __device__ __forceinline__ bf16x16 frag_B(const bf16* colp, int kbase, int half) {
  bf16x8 lo = load8(colp + kbase + half * 16);
  bf16x8 hi = load8(colp + kbase + half * 16 + 8);
  bf16x16 r;
#pragma unroll
  for (int i = 0; i < 8; ++i) { r[i] = lo[i]; r[8 + i] = hi[i]; }
  return r;
}

static __device__ __forceinline__ f32x8 wmma_bf16(bf16x16 a, bf16x16 b, f32x8 c) {
  return __builtin_amdgcn_wmma_f32_16x16x32_bf16(false, a, false, b, (short)0, c, false, false);
}

// Tensor Data Mover: DMA a (rows x tile_w) bf16 tile (row stride = `stride` elems)
// from global into LDS at byte offset lds_off. D# layout per CDNA5 ISA ch.8.
// 6-arg builtin form: (g0, g1, g2, g3, g_extra, cpol)
static __device__ __forceinline__ void tdm_load_tile(unsigned lds_off, const bf16* gaddr,
                                                     unsigned tensor_dim0, unsigned rows,
                                                     unsigned tile_w, unsigned stride) {
  unsigned long long ga = (unsigned long long)(size_t)gaddr;
  u32x4 g0;
  g0[0] = 1u;                                              // count=1, user mode
  g0[1] = lds_off;                                         // lds_addr (bytes)
  g0[2] = (unsigned)(ga & 0xffffffffu);                    // global_addr[31:0]
  g0[3] = (unsigned)((ga >> 32) & 0x1ffffffu) | (2u << 30);// addr[56:32] | type=2
  i32x8 g1;
  g1[0] = (int)(1u << 16);                                 // data_size=1 (2 bytes)
  g1[1] = (int)(tensor_dim0 << 16);                        // tensor_dim0[15:0] @bit48
  g1[2] = (int)((tensor_dim0 >> 16) | (rows << 16));       // dim0 hi | tensor_dim1 lo
  g1[3] = (int)(tile_w << 16);                             // tensor_dim1 hi | tile_dim0
  g1[4] = (int)rows;                                       // tile_dim1 (tile_dim2=0)
  g1[5] = (int)stride;                                     // tensor_dim0_stride lo32
  g1[6] = 0;                                               // stride hi | dim1_stride lo
  g1[7] = 0;
  i32x4 z4 = {0, 0, 0, 0};
  i32x8 z8 = {0, 0, 0, 0, 0, 0, 0, 0};
  __builtin_amdgcn_tensor_load_to_lds(g0, g1, z4, z4, z8, 0);
}

// ---------------- conversion / packing kernels ----------------

__global__ void f32_to_bf16_k(const float* __restrict__ src, bf16* __restrict__ dst, int n) {
  int i = blockIdx.x * blockDim.x + threadIdx.x;
  if (i < n) dst[i] = (bf16)src[i];
}

__global__ void transpose_w_k(const float* __restrict__ w, bf16* __restrict__ wt) {
  // w: K x N (1024x1024) row-major  ->  wt: N x K
  int i = blockIdx.x * blockDim.x + threadIdx.x;
  int n = i & 1023, k = i >> 10;
  wt[(size_t)n * DIM_ + k] = (bf16)w[i];
}

__global__ void concat_adapter_k(const float* __restrict__ ha, const float* __restrict__ p,
                                 bf16* __restrict__ dst) {
  int i = blockIdx.x * blockDim.x + threadIdx.x; // B*320*1024
  if (i >= B_ * KA_ * DIM_) return;
  int c = i & 1023;
  int r = (i >> 10) % KA_;
  int b = i / (KA_ * DIM_);
  float v = (r < 256) ? ha[((size_t)(b * 256 + r)) * DIM_ + c]
                      : p[((size_t)(b * 64 + (r - 256))) * DIM_ + c];
  dst[i] = (bf16)v;
}

// RoPE + repack (B,L,DIM) fp32 -> (B,H,ktot,64) bf16 at [b][h][segoff+t][d]
__global__ void rope_pack_k(const float* __restrict__ src, bf16* __restrict__ dst,
                            int L, int segoff, int ktot) {
  int gid = blockIdx.x * blockDim.x + threadIdx.x;  // B*L*NH*32 pairs
  if (gid >= B_ * L * NH_ * 32) return;
  int i = gid & 31;                    // pair -> d = 2i, 2i+1
  int h = (gid >> 5) & 15;
  int t = (gid >> 9) % L;
  int b = gid / (L * NH_ * 32);
  const float* sp = src + ((size_t)(b * L + t)) * DIM_ + h * HD_;
  float x0 = sp[2 * i], x1 = sp[2 * i + 1];
  int j0 = (2 * i) & 31, j1 = (2 * i + 1) & 31;
  float f0 = (float)t * __powf(10000.f, -(float)j0 / 32.f);
  float f1 = (float)t * __powf(10000.f, -(float)j1 / 32.f);
  float o0 = x0 * __cosf(f0) - x1 * __sinf(f0);
  float o1 = x1 * __cosf(f1) + x0 * __sinf(f1);
  bf16* dp = dst + (((size_t)(b * NH_ + h) * ktot) + segoff + t) * HD_;
  dp[2 * i] = (bf16)o0;
  dp[2 * i + 1] = (bf16)o1;
}

// V repack with transpose: (B,L,DIM) fp32 -> (B,H,64,KTOT) bf16 at [b][h][d][segoff+t]
__global__ void vpack_k(const float* __restrict__ src, bf16* __restrict__ dst,
                        int L, int segoff) {
  int gid = blockIdx.x * blockDim.x + threadIdx.x; // B*L*DIM
  if (gid >= B_ * L * DIM_) return;
  int c = gid & 1023;
  int d = c & 63, h = c >> 6;
  int t = (gid >> 10) % L;
  int b = gid / (L * DIM_);
  dst[(((size_t)(b * NH_ + h) * HD_) + d) * KTOT_ + segoff + t] = (bf16)src[gid];
}

// ---------------- WMMA GEMM: out = A(MxK,bf16) * Wt(NxK,bf16)^T + bias ----------------
// Block = 8 waves sharing one 16-row A tile; A is staged through LDS by the
// Tensor Data Mover (double buffered, TENSORcnt-tracked). B streams from L2.

#define KC_ 64   // K-chunk staged per TDM transfer (16x64 bf16 = 2KB)

__global__ __launch_bounds__(256) void gemm_bf16_k(
    const bf16* __restrict__ A, const bf16* __restrict__ Wt,
    const float* __restrict__ bias, float* __restrict__ out,
    int M, int N, int K, int relu) {
  __shared__ bf16 atile[2][16 * KC_];
  int wid  = threadIdx.x >> 5;
  int lane = threadIdx.x & 31;
  int half = lane >> 4;
  int lr   = lane & 15;
  int m0 = blockIdx.x * 16;
  int n0 = blockIdx.y * 512 + wid * 64;
  const bf16* arow = A + (size_t)m0 * K;     // block-shared 16-row A panel
  bool leader = (wid == 0);
  unsigned lds_off[2] = { (unsigned)(size_t)(&atile[0][0]),
                          (unsigned)(size_t)(&atile[1][0]) };
  int nchunk = K / KC_;

  if (leader) {
    tdm_load_tile(lds_off[0], arow, (unsigned)K, 16u, KC_, (unsigned)K);
    __builtin_amdgcn_s_wait_tensorcnt(0);
  }
  __syncthreads();

  f32x8 acc[4] = {};
  for (int c = 0; c < nchunk; ++c) {
    int buf = c & 1;
    if (leader && (c + 1 < nchunk))          // prefetch next A chunk via TDM
      tdm_load_tile(lds_off[buf ^ 1], arow + (size_t)(c + 1) * KC_,
                    (unsigned)K, 16u, KC_, (unsigned)K);
#pragma unroll
    for (int t = 0; t < 4; ++t)              // L2 prefetch hint for next W chunk
      __builtin_prefetch(Wt + (size_t)(n0 + t * 16 + lr) * K + (c + 1) * KC_, 0, 1);
#pragma unroll
    for (int ks = 0; ks < KC_; ks += 32) {
      bf16x16 af = frag_A(&atile[buf][lr * KC_], ks, half);   // ds_load_b128 x2
      int k0 = c * KC_ + ks;
#pragma unroll
      for (int t = 0; t < 4; ++t) {
        const bf16* bcol = Wt + (size_t)(n0 + t * 16 + lr) * K;
        acc[t] = wmma_bf16(af, frag_B(bcol, k0, half), acc[t]);
      }
    }
    if (leader && (c + 1 < nchunk))
      __builtin_amdgcn_s_wait_tensorcnt(0);
    __syncthreads();
  }

#pragma unroll
  for (int t = 0; t < 4; ++t) {
    int col = n0 + t * 16 + lr;
    float bv = bias[col];
#pragma unroll
    for (int j = 0; j < 8; ++j) {
      int row = m0 + half * 8 + j;
      float v = acc[t][j] + bv;
      if (relu) v = fmaxf(v, 0.f);
      out[(size_t)row * N + col] = v;
    }
  }
}

// ---------------- fused attention: one wave per 16-query tile ----------------

__global__ __launch_bounds__(256) void attn_k(
    const bf16* __restrict__ Qb,   // (B,H,T,64)
    const bf16* __restrict__ Kc,   // (B,H,KTOT,64)
    const bf16* __restrict__ Vt,   // (B,H,64,KTOT)
    const float* __restrict__ gate,
    bf16* __restrict__ Ao) {       // (B,T,DIM) bf16
  int wid  = blockIdx.x * 8 + (threadIdx.x >> 5);
  int lane = threadIdx.x & 31;
  int half = lane >> 4, lr = lane & 15;
  int qt = wid & 63;               // T/16 tiles
  int h  = (wid >> 6) & 15;
  int b  = wid >> 10;
  float rg = tanhf(gate[0]);

  const bf16* qbase = Qb + (((size_t)(b * NH_ + h) * T_) + qt * 16 + lr) * HD_;
  bf16x16 qf0 = frag_B(qbase, 0, half);    // Q^T as B-matrix, d 0..31
  bf16x16 qf1 = frag_B(qbase, 32, half);   // d 32..63
  const bf16* kbase = Kc + ((size_t)(b * NH_ + h) * KTOT_) * HD_;
  const bf16* vbase = Vt + ((size_t)(b * NH_ + h) * HD_) * KTOT_;

  f32x8 o[4] = {};
  float m = -1e30f, l = 0.f;

  for (int k0 = 0; k0 < KTOT_; k0 += 32) {
    float sc1 = 0.125f * ((k0      >= SEG_TSK) ? rg : 1.f);
    float sc2 = 0.125f * ((k0 + 16 >= SEG_TSK) ? rg : 1.f);
    const bf16* k1 = kbase + (size_t)(k0 + lr) * HD_;
    const bf16* k2 = kbase + (size_t)(k0 + 16 + lr) * HD_;
    f32x8 s1 = {}, s2 = {};
    s1 = wmma_bf16(frag_A(k1, 0, half), qf0, s1);   // S1^T = K1 * Q^T
    s1 = wmma_bf16(frag_A(k1, 32, half), qf1, s1);
    s2 = wmma_bf16(frag_A(k2, 0, half), qf0, s2);   // S2^T = K2 * Q^T
    s2 = wmma_bf16(frag_A(k2, 32, half), qf1, s2);

    float mx = -1e30f;
#pragma unroll
    for (int j = 0; j < 8; ++j) {
      s1[j] *= sc1; s2[j] *= sc2;
      mx = fmaxf(mx, fmaxf(s1[j], s2[j]));
    }
    mx = fmaxf(mx, __shfl_xor(mx, 16));
    float mn = fmaxf(m, mx);
    float alpha = __expf(m - mn);
    float ssum = 0.f;
#pragma unroll
    for (int j = 0; j < 8; ++j) {
      s1[j] = __expf(s1[j] - mn);
      s2[j] = __expf(s2[j] - mn);
      ssum += s1[j] + s2[j];
    }
    ssum += __shfl_xor(ssum, 16);
    l = l * alpha + ssum;
    m = mn;
#pragma unroll
    for (int t = 0; t < 4; ++t)
#pragma unroll
      for (int j = 0; j < 8; ++j) o[t][j] *= alpha;

    // Repack P^T (C-layout) into a B-fragment: element e <-> key half*16+e.
    bf16x16 pf;
#pragma unroll
    for (int j = 0; j < 8; ++j) {
      float x1 = __shfl_xor(s1[j], 16);
      float x2 = __shfl_xor(s2[j], 16);
      pf[j]     = (bf16)(half ? x2    : s1[j]);
      pf[8 + j] = (bf16)(half ? s2[j] : x1);
    }
    // O^T += V^T * P^T  (4 d-tiles of 16)
#pragma unroll
    for (int t = 0; t < 4; ++t) {
      const bf16* vrow = vbase + (size_t)(t * 16 + lr) * KTOT_ + k0;
      o[t] = wmma_bf16(frag_A(vrow, 0, half), pf, o[t]);
    }
  }

  float inv_l = 1.f / l;
  bf16* outp = Ao + ((size_t)(b * T_ + qt * 16 + lr)) * DIM_ + h * HD_;
#pragma unroll
  for (int t = 0; t < 4; ++t)
#pragma unroll
    for (int j = 0; j < 8; ++j)
      outp[t * 16 + half * 8 + j] = (bf16)(o[t][j] * inv_l);
}

// ---------------- residual + LayerNorm (one wave per row) ----------------

__global__ __launch_bounds__(256) void resid_ln_k(
    const float* __restrict__ proj, const float* __restrict__ x,
    const float* __restrict__ g, const float* __restrict__ beta,
    bf16* __restrict__ yln) {
  int row  = blockIdx.x * 8 + (threadIdx.x >> 5);
  int lane = threadIdx.x & 31;
  const float* pp = proj + (size_t)row * DIM_;
  const float* xp = x + (size_t)row * DIM_;
  float v[32];
  float s = 0.f;
#pragma unroll
  for (int j = 0; j < 32; ++j) { v[j] = pp[lane + 32 * j] + xp[lane + 32 * j]; s += v[j]; }
#pragma unroll
  for (int o = 16; o >= 1; o >>= 1) s += __shfl_xor(s, o);
  float mu = s * (1.f / (float)DIM_);
  float var = 0.f;
#pragma unroll
  for (int j = 0; j < 32; ++j) { float d = v[j] - mu; var += d * d; }
#pragma unroll
  for (int o = 16; o >= 1; o >>= 1) var += __shfl_xor(var, o);
  float inv = rsqrtf(var * (1.f / (float)DIM_) + 1e-5f);
  bf16* yp = yln + (size_t)row * DIM_;
#pragma unroll
  for (int j = 0; j < 32; ++j) {
    int c = lane + 32 * j;
    yp[c] = (bf16)((v[j] - mu) * inv * g[c] + beta[c]);
  }
}

// ---------------- launcher ----------------

extern "C" void kernel_launch(void* const* d_in, const int* in_sizes, int n_in,
                              void* d_out, int out_size, void* d_ws, size_t ws_size,
                              hipStream_t stream) {
  const float* x   = (const float*)d_in[0];
  const float* h_t = (const float*)d_in[1];
  const float* h_a = (const float*)d_in[2];
  const float* p   = (const float*)d_in[3];
  const float* w[9];
  const float* bias[9];
  for (int i = 0; i < 9; ++i) { w[i] = (const float*)d_in[4 + 2 * i]; bias[i] = (const float*)d_in[5 + 2 * i]; }
  const float* ln_g = (const float*)d_in[22];
  const float* ln_b = (const float*)d_in[23];
  const float* gate = (const float*)d_in[24];

  char* ws = (char*)d_ws;
  size_t off = 0;
  auto alloc = [&](size_t bytes) -> void* {
    void* r = ws + off;
    off += (bytes + 255) & ~(size_t)255;
    return r;
  };

  const int MT = B_ * T_;          // 4096 token rows
  const int MA = B_ * KA_;         // 1280 adapter rows
  const int MK = B_ * KT_;         // 2048 task rows

  bf16* wt[9];
  for (int i = 0; i < 9; ++i) wt[i] = (bf16*)alloc((size_t)DIM_ * DIM_ * 2);
  bf16* xb  = (bf16*)alloc((size_t)MT * DIM_ * 2);
  bf16* hab = (bf16*)alloc((size_t)MA * DIM_ * 2);
  bf16* htb = (bf16*)alloc((size_t)MK * DIM_ * 2);
  float* qf    = (float*)alloc((size_t)MT * DIM_ * 4);
  float* ktokf = (float*)alloc((size_t)MT * DIM_ * 4);
  float* vtokf = (float*)alloc((size_t)MT * DIM_ * 4);
  float* kadaf = (float*)alloc((size_t)MA * DIM_ * 4);
  float* vadaf = (float*)alloc((size_t)MA * DIM_ * 4);
  float* ktskf = (float*)alloc((size_t)MK * DIM_ * 4);
  float* vtskf = (float*)alloc((size_t)MK * DIM_ * 4);
  bf16* qb = (bf16*)alloc((size_t)B_ * NH_ * T_ * HD_ * 2);
  bf16* kc = (bf16*)alloc((size_t)B_ * NH_ * KTOT_ * HD_ * 2);
  bf16* vt = (bf16*)alloc((size_t)B_ * NH_ * HD_ * KTOT_ * 2);
  // tail buffers alias dead fp32 projection buffers
  bf16*  aout  = (bf16*)qf;      // needed after rope_pack consumes qf
  float* projo = ktokf;          // needed after rope_pack consumes ktokf
  bf16*  yln   = (bf16*)vtokf;   // needed after vpack consumes vtokf

  const int TB = 256;
  // weight transposes + input conversions
  for (int i = 0; i < 9; ++i)
    transpose_w_k<<<(DIM_ * DIM_) / TB, TB, 0, stream>>>(w[i], wt[i]);
  f32_to_bf16_k<<<(MT * DIM_) / TB, TB, 0, stream>>>(x, xb, MT * DIM_);
  concat_adapter_k<<<(MA * DIM_ + TB - 1) / TB, TB, 0, stream>>>(h_a, p, hab);
  f32_to_bf16_k<<<(MK * DIM_) / TB, TB, 0, stream>>>(h_t, htb, MK * DIM_);

  // 7 projections
  gemm_bf16_k<<<dim3(MT / 16, 2), TB, 0, stream>>>(xb,  wt[0], bias[0], qf,    MT, DIM_, DIM_, 0);
  gemm_bf16_k<<<dim3(MT / 16, 2), TB, 0, stream>>>(xb,  wt[1], bias[1], ktokf, MT, DIM_, DIM_, 0);
  gemm_bf16_k<<<dim3(MT / 16, 2), TB, 0, stream>>>(xb,  wt[2], bias[2], vtokf, MT, DIM_, DIM_, 0);
  gemm_bf16_k<<<dim3(MA / 16, 2), TB, 0, stream>>>(hab, wt[3], bias[3], kadaf, MA, DIM_, DIM_, 0);
  gemm_bf16_k<<<dim3(MA / 16, 2), TB, 0, stream>>>(hab, wt[4], bias[4], vadaf, MA, DIM_, DIM_, 0);
  gemm_bf16_k<<<dim3(MK / 16, 2), TB, 0, stream>>>(htb, wt[5], bias[5], ktskf, MK, DIM_, DIM_, 0);
  gemm_bf16_k<<<dim3(MK / 16, 2), TB, 0, stream>>>(htb, wt[6], bias[6], vtskf, MK, DIM_, DIM_, 0);

  // RoPE + pack Q / K, transpose-pack V
  rope_pack_k<<<(B_ * T_ * NH_ * 32 + TB - 1) / TB, TB, 0, stream>>>(qf,    qb, T_,  0,       T_);
  rope_pack_k<<<(B_ * T_ * NH_ * 32 + TB - 1) / TB, TB, 0, stream>>>(ktokf, kc, T_,  0,       KTOT_);
  rope_pack_k<<<(B_ * KA_ * NH_ * 32 + TB - 1) / TB, TB, 0, stream>>>(kadaf, kc, KA_, SEG_ADA, KTOT_);
  rope_pack_k<<<(B_ * KT_ * NH_ * 32 + TB - 1) / TB, TB, 0, stream>>>(ktskf, kc, KT_, SEG_TSK, KTOT_);
  vpack_k<<<(MT * DIM_ + TB - 1) / TB, TB, 0, stream>>>(vtokf, vt, T_,  0);
  vpack_k<<<(MA * DIM_ + TB - 1) / TB, TB, 0, stream>>>(vadaf, vt, KA_, SEG_ADA);
  vpack_k<<<(MK * DIM_ + TB - 1) / TB, TB, 0, stream>>>(vtskf, vt, KT_, SEG_TSK);

  // fused attention: B*H*(T/16) = 4096 waves, 8 waves/block
  attn_k<<<(B_ * NH_ * (T_ / 16)) / 8, TB, 0, stream>>>(qb, kc, vt, gate, aout);

  // output projection + residual/LN + final ReLU GEMM
  gemm_bf16_k<<<dim3(MT / 16, 2), TB, 0, stream>>>(aout, wt[7], bias[7], projo, MT, DIM_, DIM_, 0);
  resid_ln_k<<<MT / 8, TB, 0, stream>>>(projo, x, ln_g, ln_b, yln);
  gemm_bf16_k<<<dim3(MT / 16, 2), TB, 0, stream>>>(yln, wt[8], bias[8], (float*)d_out, MT, DIM_, DIM_, 1);
}
